// Self_Attn_22952305230524
// MI455X (gfx1250) — matrix operand order repaired and checked by reading, not compiled, not run on hardware
//
#include <hip/hip_runtime.h>

typedef __attribute__((ext_vector_type(16))) __bf16 v16bf;
typedef __attribute__((ext_vector_type(8)))  float  v8f;
typedef __attribute__((ext_vector_type(4)))  float  v4f;

#define B_   4
#define C_   256
#define N_   4096
#define K_   3
#define C8_  32

#ifndef USE_ASYNC_LDS
#define USE_ASYNC_LDS 1
#endif

// workspace byte offsets
#define OUT_OFF 0ull               // B*C*N f32 attn accumulator   (16 MiB)
#define QP_OFF  16777216ull        // K*B*N*C8 bf16 Q A-fragments  ( 3 MiB)
#define KP_OFF  19922944ull        // B*N*C8   bf16 K B-fragments  ( 1 MiB)
#define VP_OFF  20971520ull        // K*B*N*C  bf16 V B-fragments  (24 MiB)
#define WVP_OFF 46137344ull        // 256*256  bf16 Wv A-fragments (128 KiB)
#define WFP_OFF 46268416ull        // 256*512  bf16 Wf A-fragments (256 KiB)

// ---- WMMA helpers ---------------------------------------------------------

__device__ __forceinline__ v8f wmma_bf16(v16bf a, v16bf b, v8f c) {
  return __builtin_amdgcn_wmma_f32_16x16x32_bf16(false, a, false, b,
                                                 (short)0, c, false, false);
}

// A-fragment (16x32, 16-bit): lane<16 -> row=lane, K elems {0..15};
// lane>=16 -> row=lane-16, K elems {16..31}.  Element e -> K offset:
__device__ __forceinline__ int kkA(int e, int half) {
  return e + (half << 4);
}
// B-fragment (32x16, 16-bit): lane%16 = col, K = e + 16*(lane/16).

__device__ __forceinline__ v8f v8f_zero() {
  v8f z = {0.f, 0.f, 0.f, 0.f, 0.f, 0.f, 0.f, 0.f};
  return z;
}

// ---- CDNA5 async global->LDS copy (ASYNCcnt) ------------------------------

__device__ __forceinline__ void async_ld16(unsigned lds_off, const void* gaddr) {
#if USE_ASYNC_LDS
  asm volatile("global_load_async_to_lds_b128 %0, %1, off"
               :: "v"(lds_off), "v"(gaddr) : "memory");
#else
  (void)lds_off; (void)gaddr;
#endif
}

__device__ __forceinline__ void wait_async0() {
#if USE_ASYNC_LDS
#if defined(__has_builtin)
#if __has_builtin(__builtin_amdgcn_s_wait_asynccnt)
  __builtin_amdgcn_s_wait_asynccnt(0);
#else
  asm volatile("s_wait_asynccnt 0x0" ::: "memory");
#endif
#else
  asm volatile("s_wait_asynccnt 0x0" ::: "memory");
#endif
#endif
}

// ---- Kernels --------------------------------------------------------------

__global__ void zero_kernel(float* p, size_t n) {
  size_t i = (size_t)blockIdx.x * blockDim.x + threadIdx.x;
  size_t stride = (size_t)gridDim.x * blockDim.x;
  for (; i < n; i += stride) p[i] = 0.f;
}

// Pack Wv (256x256) and Wf (256x512) into bf16 A-fragments.
__global__ void pack_w_kernel(const float* __restrict__ Wv,
                              const float* __restrict__ Wf,
                              __bf16* __restrict__ wvp,
                              __bf16* __restrict__ wfp) {
  int t = blockIdx.x * blockDim.x + threadIdx.x;
  int lane = t & 31;
  int blk  = t >> 5;                      // lane-block id
  const float* W; __bf16* dst; int kdim, mt, kc, tileidx;
  if (blk < 16 * 8) {                     // Wv: 16 m-tiles x 8 k-chunks
    W = Wv; dst = wvp; kdim = 256; kc = blk & 7; mt = blk >> 3;
    tileidx = mt * 8 + kc;
  } else {                                // Wf: 16 m-tiles x 16 k-chunks
    blk -= 128;
    if (blk >= 16 * 16) return;
    W = Wf; dst = wfp; kdim = 512; kc = blk & 15; mt = blk >> 4;
    tileidx = mt * 16 + kc;
  }
  int half = lane >> 4, col = lane & 15;
  v16bf v;
  for (int e = 0; e < 16; ++e) {
    int kk = kc * 32 + kkA(e, half);
    v[e] = (__bf16)W[(size_t)(mt * 16 + col) * kdim + kk];
  }
  *(v16bf*)(dst + ((size_t)tileidx * 32 + lane) * 16) = v;
}

// Q projection -> A-fragments; K projection -> B-fragments. (small GEMMs, VALU)
// Blocks 0..383: Q jobs (K_*B_*256 tiles * 32 lanes); blocks 384..511: K jobs.
__global__ __launch_bounds__(256) void pack_qk_kernel(
    const float* __restrict__ xf, const float* __restrict__ xb,
    const float* __restrict__ Wq, const float* __restrict__ bq,
    const float* __restrict__ Wk, const float* __restrict__ bk,
    __bf16* __restrict__ qp, __bf16* __restrict__ kp) {
  __shared__ float Wl[32 * 256];
  int tid = threadIdx.x;
  bool isQ = blockIdx.x < 384;
  const float* Wsrc = isQ ? Wq : Wk;
  for (int i = tid; i < 8192; i += 256) Wl[i] = Wsrc[i];
  __syncthreads();

  float acc[16];
  int c8idx[16];
  if (isQ) {
    int qidx = blockIdx.x * 256 + tid;
    int lane = qidx & 31, tile = (qidx >> 5) & 255;
    int b = (qidx >> 13) & 3, k = qidx >> 15;
    int half = lane >> 4, colp = lane & 15;
    int m = tile * 16 + colp;               // query position
    for (int e = 0; e < 16; ++e) { c8idx[e] = kkA(e, half); acc[e] = 0.f; }
    const float* x = xb + (size_t)(k * 4 + b) * C_ * N_ + m;
    for (int c = 0; c < 256; ++c) {
      float xv = x[(size_t)c * N_];
      for (int e = 0; e < 16; ++e) acc[e] += Wl[c8idx[e] * 256 + c] * xv;
    }
    v16bf v;
    for (int e = 0; e < 16; ++e) v[e] = (__bf16)(acc[e] + bq[c8idx[e]]);
    *(v16bf*)(qp + ((size_t)((k * 4 + b) * 256 + tile) * 32 + lane) * 16) = v;
  } else {
    int kidx = (blockIdx.x - 384) * 256 + tid;
    int lane = kidx & 31, tile = (kidx >> 5) & 255, b = kidx >> 13;
    int half = lane >> 4, colp = lane & 15;
    int j = tile * 16 + colp;               // key position
    for (int e = 0; e < 16; ++e) { c8idx[e] = e + 16 * half; acc[e] = 0.f; }
    const float* x = xf + (size_t)b * C_ * N_ + j;
    for (int c = 0; c < 256; ++c) {
      float xv = x[(size_t)c * N_];
      for (int e = 0; e < 16; ++e) acc[e] += Wl[c8idx[e] * 256 + c] * xv;
    }
    v16bf v;
    for (int e = 0; e < 16; ++e) v[e] = (__bf16)(acc[e] + bk[c8idx[e]]);
    *(v16bf*)(kp + ((size_t)(b * 256 + tile) * 32 + lane) * 16) = v;
  }
}

// V = Wv @ xb + bv via WMMA, stored as B-fragments over (key j, channel c).
// grid: (n_chunk 64, m_chunk 4, kb 12); block 128 (4 waves).
__global__ __launch_bounds__(128) void pack_v_kernel(
    const float* __restrict__ xb, const __bf16* __restrict__ wvp,
    const float* __restrict__ bv, __bf16* __restrict__ vpack) {
  __shared__ __bf16 T_lds[64 * 72];
  int tid = threadIdx.x, wave = tid >> 5, lane = tid & 31;
  int half = lane >> 4, col = lane & 15;
  int nchunk = blockIdx.x, mchunk = blockIdx.y, kb = blockIdx.z;
  int m_tile = mchunk * 4 + wave;
  const float* xsrc = xb + (size_t)kb * C_ * N_;
  int n0 = nchunk * 64;

  v8f acc[4];
  for (int nt = 0; nt < 4; ++nt) acc[nt] = v8f_zero();

  for (int kc = 0; kc < 8; ++kc) {
    v16bf af = *(const v16bf*)(wvp + ((size_t)(m_tile * 8 + kc) * 32 + lane) * 16);
    for (int nt = 0; nt < 4; ++nt) {
      v16bf bfr;
      for (int e = 0; e < 16; ++e) {
        int c = kc * 32 + e + 16 * half;
        int n = n0 + nt * 16 + col;
        bfr[e] = (__bf16)xsrc[(size_t)c * N_ + n];
      }
      acc[nt] = wmma_bf16(af, bfr, acc[nt]);
    }
  }
  // bias + transpose-stage (channel-major) in LDS
  for (int nt = 0; nt < 4; ++nt)
    for (int r = 0; r < 8; ++r) {
      int m  = m_tile * 16 + r + 8 * half;    // out channel (global)
      int lc = wave * 16 + r + 8 * half;      // local channel
      int ln = nt * 16 + col;                 // local position
      T_lds[lc * 72 + ln] = (__bf16)(acc[nt][r] + bv[m]);
    }
  __syncthreads();
  // emit V B-fragments: tile (j_chunk of 32 keys, c_tile of 16 channels)
  __bf16* vkb = vpack + (size_t)kb * N_ * C_;
  for (int task = tid; task < 256; task += 128) {
    int tl  = task & 31;
    int ctl = (task >> 5) & 3;
    int jcl = task >> 7;
    int th = tl >> 4, tc = tl & 15;
    v16bf v;
    for (int e = 0; e < 16; ++e) {
      int lc = ctl * 16 + tc;
      int ln = jcl * 32 + e + 16 * th;
      v[e] = T_lds[lc * 72 + ln];
    }
    int jc  = nchunk * 2 + jcl;
    int ctg = mchunk * 4 + ctl;
    *(v16bf*)(vkb + ((size_t)(jc * 16 + ctg) * 32 + tl) * 16) = v;
  }
}

// Flash attention: grid (i_chunk 64, b 4, k 3); block 128 (4 waves).
// Wave owns 16 queries x 256 channels; loop over 128 key-blocks of 32.
// V fragments are staged into LDS with double-buffered async copies.
__global__ __launch_bounds__(128) void attn_kernel(
    const __bf16* __restrict__ qp, const __bf16* __restrict__ kp,
    const __bf16* __restrict__ vp, float* __restrict__ outs) {
  __shared__ float  S_lds[4][16][33];
  __shared__ __bf16 P_lds[4][16][32];
  __shared__ float  mrow[4][16], lrow[4][16], arow[4][16];
  __shared__ __align__(32) __bf16 V_lds[2][8192];  // 2 x (32 keys x 256 ch)

  int tid = threadIdx.x, wave = tid >> 5, lane = tid & 31;
  int half = lane >> 4, col = lane & 15;
  int ic = blockIdx.x, b = blockIdx.y, k = blockIdx.z;
  int i_tile = ic * 4 + wave;

  v16bf qfrag = *(const v16bf*)(qp +
      ((size_t)((k * B_ + b) * 256 + i_tile) * 32 + lane) * 16);
  const __bf16* kpb = kp + (size_t)b * 256 * 32 * 16;
  const __bf16* vkb = vp + (size_t)(k * B_ + b) * N_ * C_;

  // per-thread async stage of one 16 KiB key-block (8 x b128 per thread)
  unsigned lds_base[2];
  lds_base[0] = (unsigned)(size_t)(&V_lds[0][0]);
  lds_base[1] = (unsigned)(size_t)(&V_lds[1][0]);
#if !USE_ASYNC_LDS
  v4f* dl0 = (v4f*)&V_lds[0][0];
  v4f* dl1 = (v4f*)&V_lds[1][0];
#endif

  v8f acc[16];
  for (int ct = 0; ct < 16; ++ct) acc[ct] = v8f_zero();
  if (lane < 16) { mrow[wave][lane] = -1e30f; lrow[wave][lane] = 0.f; }

  // prologue: stage block 0 into buffer 0
#if USE_ASYNC_LDS
  {
    const char* g0 = (const char*)(vkb);
    for (int t = 0; t < 8; ++t) {
      int i = tid + 128 * t;
      async_ld16(lds_base[0] + i * 16, g0 + (size_t)i * 16);
    }
  }
#else
  {
    const v4f* src = (const v4f*)(vkb);
    for (int i = tid; i < 1024; i += 128) dl0[i] = src[i];
  }
#endif

  for (int jc = 0; jc < 128; ++jc) {
    int cur = jc & 1;
    wait_async0();          // own async copies into buffer `cur` done
    __syncthreads();        // everyone's copies visible; prev readers done

    // kick off the next block's stage while we compute on `cur`
    if (jc + 1 < 128) {
#if USE_ASYNC_LDS
      const char* gn = (const char*)(vkb + (size_t)(jc + 1) * 8192);
      for (int t = 0; t < 8; ++t) {
        int i = tid + 128 * t;
        async_ld16(lds_base[cur ^ 1] + i * 16, gn + (size_t)i * 16);
      }
#else
      const v4f* src = (const v4f*)(vkb + (size_t)(jc + 1) * 8192);
      v4f* dln = cur ? dl0 : dl1;
      for (int i = tid; i < 1024; i += 128) dln[i] = src[i];
#endif
    }

    // S = Q (16x32) x K (32x32 keys) -> two 16x16 tiles
    v16bf kf0 = *(const v16bf*)(kpb + ((size_t)(jc * 2 + 0) * 32 + lane) * 16);
    v16bf kf1 = *(const v16bf*)(kpb + ((size_t)(jc * 2 + 1) * 32 + lane) * 16);
    v8f s0 = wmma_bf16(qfrag, kf0, v8f_zero());
    v8f s1 = wmma_bf16(qfrag, kf1, v8f_zero());
    for (int r = 0; r < 8; ++r) {
      S_lds[wave][r + 8 * half][col]      = s0[r];
      S_lds[wave][r + 8 * half][16 + col] = s1[r];
    }
    __syncthreads();

    // online softmax: lanes 0..15 each own one query row
    if (lane < 16) {
      int row = lane;
      float m0 = mrow[wave][row];
      float mx = -1e30f;
      for (int j = 0; j < 32; ++j) mx = fmaxf(mx, S_lds[wave][row][j]);
      float mnew = fmaxf(m0, mx);
      float al = __expf(m0 - mnew);
      float sum = 0.f;
      for (int j = 0; j < 32; ++j) {
        float p = __expf(S_lds[wave][row][j] - mnew);
        sum += p;
        P_lds[wave][row][j] = (__bf16)p;
      }
      mrow[wave][row] = mnew;
      lrow[wave][row] = lrow[wave][row] * al + sum;
      arow[wave][row] = al;
    }
    __syncthreads();

    v8f av;
    for (int r = 0; r < 8; ++r) av[r] = arow[wave][r + 8 * half];
    v16bf pf;
    for (int e = 0; e < 16; ++e) pf[e] = P_lds[wave][col][kkA(e, half)];

    // O = O*alpha + P (16x32) x V (32 x 16ch) over 16 channel tiles
    const __bf16* vb = &V_lds[cur][0];
    for (int ct = 0; ct < 16; ++ct) {
      v16bf vf = *(const v16bf*)(vb + (ct * 32 + lane) * 16);
      acc[ct] = wmma_bf16(pf, vf, acc[ct] * av);
    }
    // no trailing barrier: next iteration's wait+barrier publishes/reprotects
  }

  v8f linv;
  for (int r = 0; r < 8; ++r) linv[r] = 1.0f / lrow[wave][r + 8 * half];
  for (int ct = 0; ct < 16; ++ct) {
    int c = ct * 16 + col;
    v8f val = acc[ct] * linv;
    for (int r = 0; r < 8; ++r) {
      int q = i_tile * 16 + r + 8 * half;
      atomicAdd(outs + ((size_t)b * C_ + c) * N_ + q, val[r]);
    }
  }
}

// final = Wf @ [gamma*outs ; x_f] + bf via WMMA.
// grid (n_chunk 64, m_chunk 4, b 4); block 128.
__global__ __launch_bounds__(128) void final_kernel(
    const float* __restrict__ outs, const float* __restrict__ xf,
    const __bf16* __restrict__ wfp, const float* __restrict__ bfv,
    const float* __restrict__ gamma, float* __restrict__ out) {
  int tid = threadIdx.x, wave = tid >> 5, lane = tid & 31;
  int half = lane >> 4, col = lane & 15;
  int nchunk = blockIdx.x, mchunk = blockIdx.y, b = blockIdx.z;
  int m_tile = mchunk * 4 + wave;
  int n0 = nchunk * 64;
  float g = gamma[0];

  v8f acc[4];
  for (int nt = 0; nt < 4; ++nt) acc[nt] = v8f_zero();

  for (int kc = 0; kc < 16; ++kc) {
    v16bf af = *(const v16bf*)(wfp + ((size_t)(m_tile * 16 + kc) * 32 + lane) * 16);
    for (int nt = 0; nt < 4; ++nt) {
      v16bf bfr;
      for (int e = 0; e < 16; ++e) {
        int c = kc * 32 + e + 16 * half;
        int n = n0 + nt * 16 + col;
        float v = (c < 256) ? g * outs[((size_t)b * C_ + c) * N_ + n]
                            : xf[((size_t)b * C_ + (c - 256)) * N_ + n];
        bfr[e] = (__bf16)v;
      }
      acc[nt] = wmma_bf16(af, bfr, acc[nt]);
    }
  }
  for (int nt = 0; nt < 4; ++nt)
    for (int r = 0; r < 8; ++r) {
      int m = m_tile * 16 + r + 8 * half;
      int n = n0 + nt * 16 + col;
      out[((size_t)b * C_ + m) * N_ + n] = acc[nt][r] + bfv[m];
    }
}

// ---- Launch ---------------------------------------------------------------

extern "C" void kernel_launch(void* const* d_in, const int* in_sizes, int n_in,
                              void* d_out, int out_size, void* d_ws, size_t ws_size,
                              hipStream_t stream) {
  (void)in_sizes; (void)n_in; (void)out_size; (void)ws_size;
  const float* x_f = (const float*)d_in[0];
  const float* x_b = (const float*)d_in[1];
  const float* Wq  = (const float*)d_in[2];
  const float* bq  = (const float*)d_in[3];
  const float* Wk  = (const float*)d_in[4];
  const float* bk  = (const float*)d_in[5];
  const float* Wv  = (const float*)d_in[6];
  const float* bv  = (const float*)d_in[7];
  const float* Wf  = (const float*)d_in[8];
  const float* bf  = (const float*)d_in[9];
  const float* gm  = (const float*)d_in[10];
  float* out = (float*)d_out;

  char* ws = (char*)d_ws;
  float*  outs = (float*)(ws + OUT_OFF);
  __bf16* qp   = (__bf16*)(ws + QP_OFF);
  __bf16* kpk  = (__bf16*)(ws + KP_OFF);
  __bf16* vpk  = (__bf16*)(ws + VP_OFF);
  __bf16* wvp  = (__bf16*)(ws + WVP_OFF);
  __bf16* wfp  = (__bf16*)(ws + WFP_OFF);

  zero_kernel<<<2048, 256, 0, stream>>>(outs, (size_t)B_ * C_ * N_);
  pack_w_kernel<<<48, 256, 0, stream>>>(Wv, Wf, wvp, wfp);
  pack_qk_kernel<<<512, 256, 0, stream>>>(x_f, x_b, Wq, bq, Wk, bk, qp, kpk);
  pack_v_kernel<<<dim3(64, 4, 12), 128, 0, stream>>>(x_b, wvp, bv, vpk);
  attn_kernel<<<dim3(64, 4, 3), 128, 0, stream>>>(qp, kpk, vpk, outs);
  final_kernel<<<dim3(64, 4, 4), 128, 0, stream>>>(outs, x_f, wfp, bf, gm, out);
}
